// EHGNN_53953379173265
// MI455X (gfx1250) — matrix operand order repaired and checked by reading, not compiled, not run on hardware
//
#include <hip/hip_runtime.h>
#include <hip/hip_bf16.h>
#include <math.h>

typedef __bf16 bf16;
typedef bf16  v16bf  __attribute__((ext_vector_type(16)));
typedef bf16  bf16x8 __attribute__((ext_vector_type(8)));
typedef float v8f    __attribute__((ext_vector_type(8)));

#define T_ 3
#define N_ 100000
#define F_ 128
#define H_ 128
#define O_ 64
#define M_ 4
#define E_ 65536
#define B_ 65536

#define WPB 4   // waves per block
#define TPW 4   // row-tiles per wave (64 rows -> never crosses an (m,t) boundary)

// Load a B fragment: chunks at +0 and +16 elements (2 x b128).
__device__ __forceinline__ v16bf ld_frag16(const bf16* p) {
  bf16x8 lo  = *(const bf16x8*)p;
  bf16x8 hi8 = *(const bf16x8*)(p + 16);
  v16bf b;
  #pragma unroll
  for (int i = 0; i < 8; i++) { b[i] = lo[i]; b[8 + i] = hi8[i]; }
  return b;
}

// Load a layer-2 A fragment: chunks at +0 and +16 elements.
__device__ __forceinline__ v16bf ld_fragA2(const bf16* p) {
  bf16x8 lo  = *(const bf16x8*)p;
  bf16x8 hi8 = *(const bf16x8*)(p + 16);
  v16bf a;
  #pragma unroll
  for (int i = 0; i < 8; i++) { a[i] = lo[i]; a[8 + i] = hi8[i]; }
  return a;
}

// Fragment address helpers (i = flattened (ks, n) index).
__device__ __forceinline__ const bf16* w1_ptr(const bf16* sW1, int lrow, int kboff, int i) {
  return sW1 + ((i & 7) * 16 + lrow) * F_ + (i >> 3) * 32 + kboff;
}
__device__ __forceinline__ const bf16* w2_ptr(const bf16* sW2, int lrow, int kboff, int i) {
  return sW2 + ((i & 3) * 16 + lrow) * H_ + (i >> 2) * 32 + kboff;
}

// ---------------------------------------------------------------------------
// MLP tile: 16 rows x (128 -> relu 128 -> 64), bf16 WMMA, f32 accum.
// Weight fragments stream from LDS with an explicit depth-3 software pipeline
// so every s_wait_dscnt covers in-flight loads.
// A-fragment (16-bit 16x32): lane L<16: row L, K={0..7,16..23}+ks*32;
//                            lane L>=16: row L-16, K={8..15,24..31}+ks*32.
// B-fragment (16-bit 32x16): lane: col=L%16, K same as A-fragment ({0..7,16..23} / {8..15,24..31})+ks*32.
// C/D (f32 16x16): VGPR r, lane L: M=r+(L>=16?8:0), N=L%16.
// ---------------------------------------------------------------------------
__device__ __forceinline__ void mlp_tile(
    const float* __restrict__ xrow,
    const bf16*  __restrict__ sW1,          // LDS [H_][F_]
    const bf16*  __restrict__ sW2,          // LDS [O_][H_]
    const float (&b1f)[8],
    const float (&b2f)[4],
    bf16* __restrict__ st,                  // per-wave LDS staging [16*H_]
    int lane, v8f y[4], float scale[8])
{
  const int  lrow  = lane & 15;
  const bool hi    = lane >= 16;
  const int  kaoff = hi ? 8 : 0;
  const int  kboff = hi ? 8 : 0;

  // ---- build all 4 A fragments up front (16 clause-able b128 loads) ----
  v16bf a4[4];
  #pragma unroll
  for (int ks = 0; ks < 4; ks++) {
    const float4* px0 = (const float4*)(xrow + ks * 32 + kaoff);
    const float4* px1 = (const float4*)(xrow + ks * 32 + kaoff + 16);
    float4 f0 = px0[0], f1 = px0[1];
    float4 f2 = px1[0], f3 = px1[1];
    v16bf a;
    a[0]=(bf16)f0.x; a[1]=(bf16)f0.y; a[2]=(bf16)f0.z; a[3]=(bf16)f0.w;
    a[4]=(bf16)f1.x; a[5]=(bf16)f1.y; a[6]=(bf16)f1.z; a[7]=(bf16)f1.w;
    a[8]=(bf16)f2.x; a[9]=(bf16)f2.y; a[10]=(bf16)f2.z; a[11]=(bf16)f2.w;
    a[12]=(bf16)f3.x; a[13]=(bf16)f3.y; a[14]=(bf16)f3.z; a[15]=(bf16)f3.w;
    a4[ks] = a;
  }

  v8f acc[8];
  #pragma unroll
  for (int i = 0; i < 8; i++) acc[i] = {};

  // ---- layer 1: flattened 32-step WMMA stream, depth-3 B pipeline ----
  v16bf b0  = ld_frag16(w1_ptr(sW1, lrow, kboff, 0));
  v16bf b1v = ld_frag16(w1_ptr(sW1, lrow, kboff, 1));
  v16bf b2v = ld_frag16(w1_ptr(sW1, lrow, kboff, 2));
  #pragma unroll
  for (int i = 0; i < 32; i++) {
    const int ks = i >> 3, n8 = i & 7;
    v16bf bn = b2v;
    if (i + 3 < 32) bn = ld_frag16(w1_ptr(sW1, lrow, kboff, i + 3));
    acc[n8] = __builtin_amdgcn_wmma_f32_16x16x32_bf16(
        false, a4[ks], false, b0, (short)0, acc[n8], false, false);
    b0 = b1v; b1v = b2v; b2v = bn;
  }

  // ---- bias + relu -> LDS staging (bf16), inter-layer transpose ----
  #pragma unroll
  for (int n8 = 0; n8 < 8; n8++) {
    #pragma unroll
    for (int r = 0; r < 8; r++) {
      float h = fmaxf(acc[n8][r] + b1f[n8], 0.0f);
      int mr = r + (hi ? 8 : 0);
      st[mr * H_ + n8 * 16 + lrow] = (bf16)h;
    }
  }

  // ---- layer 2: A fragments from staging, B pipelined (depth 3) from sW2 ----
  v16bf a2[4];
  #pragma unroll
  for (int ks = 0; ks < 4; ks++)
    a2[ks] = ld_fragA2(st + lrow * H_ + ks * 32 + kaoff);

  #pragma unroll
  for (int i = 0; i < 4; i++) y[i] = {};

  v16bf c0  = ld_frag16(w2_ptr(sW2, lrow, kboff, 0));
  v16bf c1v = ld_frag16(w2_ptr(sW2, lrow, kboff, 1));
  v16bf c2v = ld_frag16(w2_ptr(sW2, lrow, kboff, 2));
  #pragma unroll
  for (int i = 0; i < 16; i++) {
    const int ks = i >> 2, n4 = i & 3;
    v16bf cn = c2v;
    if (i + 3 < 16) cn = ld_frag16(w2_ptr(sW2, lrow, kboff, i + 3));
    y[n4] = __builtin_amdgcn_wmma_f32_16x16x32_bf16(
        false, a2[ks], false, c0, (short)0, y[n4], false, false);
    c0 = c1v; c1v = c2v; c2v = cn;
  }

  // ---- bias + row L2 norm (16-lane butterfly per half-wave) ----
  float ssq[8];
  #pragma unroll
  for (int r = 0; r < 8; r++) ssq[r] = 0.0f;
  #pragma unroll
  for (int n4 = 0; n4 < 4; n4++) {
    #pragma unroll
    for (int r = 0; r < 8; r++) {
      float v = y[n4][r] + b2f[n4];
      y[n4][r] = v;
      ssq[r] += v * v;
    }
  }
  #pragma unroll
  for (int r = 0; r < 8; r++) {
    float s = ssq[r];
    #pragma unroll
    for (int off = 8; off >= 1; off >>= 1) s += __shfl_xor(s, off, 32);
    scale[r] = 10.0f / fmaxf(sqrtf(s), 1e-12f);
  }
}

__device__ __forceinline__ void copy_to_lds(const bf16* __restrict__ src_g,
                                            bf16* __restrict__ dst_s,
                                            int nelem, int nthreads) {
  const uint4* src = (const uint4*)src_g;
  uint4* dst = (uint4*)dst_s;
  const int n16 = (nelem * 2) / 16;
  for (int i = threadIdx.x; i < n16; i += nthreads) dst[i] = src[i];
}

__device__ __forceinline__ void prefetch_row(const float* p) {
  __builtin_prefetch(p +  0, 0, 3);
  __builtin_prefetch(p + 32, 0, 3);
  __builtin_prefetch(p + 64, 0, 3);
  __builtin_prefetch(p + 96, 0, 3);
}

// ---------------------------------------------------------------------------
__global__ void prep_kernel(const float* __restrict__ W1, const float* __restrict__ W2,
                            const float* __restrict__ mw, const float* __restrict__ nw,
                            bf16* __restrict__ W1t, bf16* __restrict__ W2t,
                            float* __restrict__ coefs) {
  int tid = threadIdx.x;
  for (int i = tid; i < H_ * F_; i += 256) {
    int n = i >> 7, k = i & 127;
    W1t[i] = (bf16)W1[(size_t)k * H_ + n];
  }
  for (int i = tid; i < O_ * H_; i += 256) {
    int o = i >> 7, k = i & 127;
    W2t[i] = (bf16)W2[(size_t)k * O_ + o];
  }
  if (tid == 0) {
    float mx = -3.4e38f;
    for (int i = 0; i < M_; i++) mx = fmaxf(mx, mw[i]);
    float s = 0.0f, e[M_];
    for (int i = 0; i < M_; i++) { e[i] = expf(mw[i] - mx); s += e[i]; }
    for (int i = 0; i < M_; i++) coefs[i] = e[i] / s;
    mx = -3.4e38f;
    for (int i = 0; i < T_; i++) mx = fmaxf(mx, nw[i]);
    s = 0.0f; float et[T_];
    for (int i = 0; i < T_; i++) { et[i] = expf(nw[i] - mx); s += et[i]; }
    for (int i = 0; i < T_; i++) coefs[4 + i] = et[i] / s;
    coefs[7] = 0.0f;
  }
}

// ---------------------------------------------------------------------------
__global__ __launch_bounds__(WPB * 32) void edge_kernel(
    const float* __restrict__ X, const int* __restrict__ fmap,
    const int* __restrict__ s_idxs, const int* __restrict__ t_idxs,
    const float* __restrict__ weightss,
    const float* __restrict__ b1, const float* __restrict__ b2,
    const bf16* __restrict__ W1t, const bf16* __restrict__ W2t,
    const float* __restrict__ coefs,
    float* __restrict__ out) {
  __shared__ __align__(16) bf16 sW1[H_ * F_];           // 32 KB
  __shared__ __align__(16) bf16 sW2[O_ * H_];           // 16 KB
  __shared__ __align__(16) bf16 stage[WPB][16 * H_];    // 16 KB

  copy_to_lds(W1t, sW1, H_ * F_, WPB * 32);
  copy_to_lds(W2t, sW2, O_ * H_, WPB * 32);

  const int lane = threadIdx.x & 31;
  const int wave = threadIdx.x >> 5;
  const int lrow = lane & 15;
  const bool hi  = lane >= 16;

  // loop invariants for this wave's 4 tiles (same (m,t) for all of them)
  const int tile0   = (blockIdx.x * WPB + wave) * TPW;
  const int rowb0   = tile0 * 16;
  const int m       = rowb0 / (T_ * E_);
  const int t       = (rowb0 / E_) % T_;
  const int ebase0  = rowb0 % E_;
  const int idxbase = (m * T_ + t) * E_;
  const float wbase = 0.5f * coefs[m] * coefs[4 + t];

  float b1f[8], b2f[4];
  #pragma unroll
  for (int n8 = 0; n8 < 8; n8++) b1f[n8] = b1[n8 * 16 + lrow];
  #pragma unroll
  for (int n4 = 0; n4 < 4; n4++) b2f[n4] = b2[n4 * 16 + lrow];

  // resolve all gather chains up front; keep OFFSETS so loads stay global_*
  size_t xoff[TPW];
  #pragma unroll
  for (int i = 0; i < TPW; i++) {
    int tid = t_idxs[idxbase + ebase0 + i * 16 + lrow];
    int mi  = fmap[t * N_ + tid];
    xoff[i] = ((size_t)t * N_ + mi) * F_;
  }

  __syncthreads();
  bf16* st = stage[wave];
  prefetch_row(X + xoff[0]);

  for (int tt = 0; tt < TPW; tt++) {
    if (tt + 1 < TPW) prefetch_row(X + xoff[tt + 1]);

    v8f y[4];
    float scale[8];
    mlp_tile(X + xoff[tt], sW1, sW2, b1f, b2f, st, lane, y, scale);

    // scatter: (1-ALPHA) * nm[m] * nt[t] * w_e * y_normalized
    const int ebase = ebase0 + tt * 16;
    #pragma unroll
    for (int r = 0; r < 8; r++) {
      const int mr = r + (hi ? 8 : 0);
      const int ei = idxbase + ebase + mr;
      const float w = weightss[ei] * wbase * scale[r];
      const int sid = s_idxs[ei];
      float* orow = out + (size_t)sid * O_;
      #pragma unroll
      for (int n4 = 0; n4 < 4; n4++) {
        atomicAdd(orow + n4 * 16 + lrow, y[n4][r] * w);
      }
    }
  }
}

// ---------------------------------------------------------------------------
__global__ __launch_bounds__(WPB * 32) void finalize_kernel(
    const float* __restrict__ s_features,
    const float* __restrict__ b1, const float* __restrict__ b2,
    const bf16* __restrict__ W1t, const bf16* __restrict__ W2t,
    float* __restrict__ out) {
  __shared__ __align__(16) bf16 sW1[H_ * F_];
  __shared__ __align__(16) bf16 sW2[O_ * H_];
  __shared__ __align__(16) bf16 stage[WPB][16 * H_];

  copy_to_lds(W1t, sW1, H_ * F_, WPB * 32);
  copy_to_lds(W2t, sW2, O_ * H_, WPB * 32);

  const int lane = threadIdx.x & 31;
  const int wave = threadIdx.x >> 5;
  const int lrow = lane & 15;
  const bool hi  = lane >= 16;

  float b1f[8], b2f[4];
  #pragma unroll
  for (int n8 = 0; n8 < 8; n8++) b1f[n8] = b1[n8 * 16 + lrow];
  #pragma unroll
  for (int n4 = 0; n4 < 4; n4++) b2f[n4] = b2[n4 * 16 + lrow];

  const int tile0 = (blockIdx.x * WPB + wave) * TPW;
  __syncthreads();
  bf16* st = stage[wave];

  for (int tt = 0; tt < TPW; tt++) {
    const int rowbase = (tile0 + tt) * 16;
    const float* xrow = s_features + (size_t)(rowbase + lrow) * F_;
    if (tt + 1 < TPW)
      prefetch_row(s_features + (size_t)(rowbase + 16 + lrow) * F_);

    v8f y[4];
    float scale[8];
    mlp_tile(xrow, sW1, sW2, b1f, b2f, st, lane, y, scale);

    // blend with scatter accumulator and apply row log-softmax in place
    #pragma unroll
    for (int r = 0; r < 8; r++) {
      const int mr = r + (hi ? 8 : 0);
      float* orow = out + (size_t)(rowbase + mr) * O_;
      float v[4];
      float mx = -3.4e38f;
      #pragma unroll
      for (int n4 = 0; n4 < 4; n4++) {
        v[n4] = orow[n4 * 16 + lrow] + 0.5f * (y[n4][r] * scale[r]);
        mx = fmaxf(mx, v[n4]);
      }
      #pragma unroll
      for (int off = 8; off >= 1; off >>= 1) mx = fmaxf(mx, __shfl_xor(mx, off, 32));
      float s = 0.0f;
      #pragma unroll
      for (int n4 = 0; n4 < 4; n4++) s += expf(v[n4] - mx);
      #pragma unroll
      for (int off = 8; off >= 1; off >>= 1) s += __shfl_xor(s, off, 32);
      const float sub = mx + logf(s);
      #pragma unroll
      for (int n4 = 0; n4 < 4; n4++) orow[n4 * 16 + lrow] = v[n4] - sub;
    }
  }
}

// ---------------------------------------------------------------------------
extern "C" void kernel_launch(void* const* d_in, const int* in_sizes, int n_in,
                              void* d_out, int out_size, void* d_ws, size_t ws_size,
                              hipStream_t stream) {
  (void)in_sizes; (void)n_in; (void)ws_size;
  const float* X           = (const float*)d_in[0];
  const float* s_features  = (const float*)d_in[1];
  const int*   fmap        = (const int*)d_in[2];
  const int*   s_idxs      = (const int*)d_in[3];
  const int*   t_idxs      = (const int*)d_in[4];
  const float* weightss    = (const float*)d_in[5];
  // d_in[6] = batch_size (compile-time B_ = 65536)
  const float* W1          = (const float*)d_in[7];
  const float* b1          = (const float*)d_in[8];
  const float* W2          = (const float*)d_in[9];
  const float* b2          = (const float*)d_in[10];
  const float* mw          = (const float*)d_in[11];
  const float* nw          = (const float*)d_in[12];
  float* out = (float*)d_out;

  bf16*  W1t   = (bf16*)d_ws;                          // 32768 B
  bf16*  W2t   = (bf16*)((char*)d_ws + 32768);         // 16384 B
  float* coefs = (float*)((char*)d_ws + 49152);        // 32 B

  prep_kernel<<<1, 256, 0, stream>>>(W1, W2, mw, nw, W1t, W2t, coefs);
  hipMemsetAsync(d_out, 0, (size_t)out_size * sizeof(float), stream);

  const int edge_tiles = (M_ * T_ * E_) / 16;          // 49152
  edge_kernel<<<edge_tiles / (WPB * TPW), WPB * 32, 0, stream>>>(
      X, fmap, s_idxs, t_idxs, weightss, b1, b2, W1t, W2t, coefs, out);

  const int self_tiles = B_ / 16;                      // 4096
  finalize_kernel<<<self_tiles / (WPB * TPW), WPB * 32, 0, stream>>>(
      s_features, b1, b2, W1t, W2t, out);
}
